// GraphIsomorphismLayer_17171279249896
// MI455X (gfx1250) — compile-verified, matrix operand order checked by reading, not run on hardware
//
#include <hip/hip_runtime.h>
#include <hip/hip_bf16.h>

typedef __bf16 bf16;
typedef __attribute__((ext_vector_type(8)))  bf16  bf16x8;
typedef __attribute__((ext_vector_type(16))) bf16  v16bf;
typedef __attribute__((ext_vector_type(8)))  float v8f;
typedef __attribute__((ext_vector_type(4)))  float f32x4;
typedef __attribute__((ext_vector_type(4)))  unsigned int u32x4;
typedef __attribute__((ext_vector_type(8)))  int   i32x8;
typedef __attribute__((ext_vector_type(4)))  int   i32x4;

#define DIM    256
#define LDSP   264      // padded LDS pitch in bf16 elements (= 132 DWORDs/row)
#define BNEPS  1e-5f

// ---------------------------------------------------------------------------
// K0: agg = eps_w * x (fp32), x_bf = bf16(x), w1b/w2b = bf16(w1/w2),
//     zero the 1024-float stats block
// ---------------------------------------------------------------------------
__global__ __launch_bounds__(256) void k0_init(
    const float* __restrict__ x, const float* __restrict__ epsw,
    const float* __restrict__ w1, const float* __restrict__ w2,
    float* __restrict__ agg, bf16* __restrict__ xb,
    bf16* __restrict__ w1b, bf16* __restrict__ w2b,
    float* __restrict__ stats, int total8)
{
    int t = blockIdx.x * blockDim.x + threadIdx.x;
    if (t < 1024) stats[t] = 0.0f;
    if (t < 16384) {                       // weight conversion: 2 x 8192 groups
        const float* wsrc = (t < 8192) ? w1 : w2;
        bf16*        wdst = (t < 8192) ? w1b : w2b;
        int g = t & 8191;
        const f32x4* wp = (const f32x4*)wsrc + (size_t)g * 2;
        f32x4 a = wp[0], b = wp[1];
        bf16x8 wv;
#pragma unroll
        for (int i = 0; i < 4; ++i) { wv[i] = (bf16)a[i]; wv[4 + i] = (bf16)b[i]; }
        ((bf16x8*)wdst)[g] = wv;
    }
    if (t >= total8) return;
    float e = epsw[0];
    const f32x4* xp = (const f32x4*)x + (size_t)t * 2;
    f32x4 a = xp[0], b = xp[1];
    f32x4 ea, eb;
    bf16x8 bv;
#pragma unroll
    for (int i = 0; i < 4; ++i) {
        ea[i] = e * a[i];
        eb[i] = e * b[i];
        bv[i]     = (bf16)a[i];
        bv[i + 4] = (bf16)b[i];
    }
    f32x4* ap = (f32x4*)agg + (size_t)t * 2;
    ap[0] = ea; ap[1] = eb;
    ((bf16x8*)xb)[t] = bv;
}

// ---------------------------------------------------------------------------
// K1: edge scatter-add. One wave per edge: 32 lanes x 8 elements.
// Gather bf16 row (L2-resident), atomic-add fp32 into agg (L2-resident).
// ---------------------------------------------------------------------------
__global__ __launch_bounds__(256) void k1_edges(
    const int* __restrict__ esrc, const int* __restrict__ edst,
    const float* __restrict__ evals, const bf16* __restrict__ xb,
    float* __restrict__ agg, int E)
{
    int eidx = blockIdx.x * 8 + (threadIdx.x >> 5);   // wave-uniform edge id
    int lane = threadIdx.x & 31;
    if (eidx >= E) return;
    int   s = esrc[eidx];
    int   d = edst[eidx];
    float v = evals[eidx];
    bf16x8 xv = *(const bf16x8*)(xb + (size_t)s * DIM + lane * 8);
    float* ar = agg + (size_t)d * DIM + lane * 8;
#pragma unroll
    for (int i = 0; i < 8; ++i)
        atomicAdd(ar + i, v * (float)xv[i]);
}

// ---------------------------------------------------------------------------
// WMMA GEMM:  Out[n,j] = rowScale[n] * sum_k A[n,k]*Wb[j,k]
// Wb (bf16 256x256 row-major) is DMA'd into LDS by the Tensor Data Mover with
// hardware row padding (pad_interval=6 -> every 128 DWORDs, pad_amount=3 ->
// 4 DWORDs) giving the bank-conflict-free 264-element pitch.
// Fused per-column sum / sum-of-squares into statSum/statSqs (atomics).
// ---------------------------------------------------------------------------
template <bool A_F32, bool OUT_F32, bool ROW_SCALE>
__global__ __launch_bounds__(256) void gemm_wmma(
    const void* __restrict__ Aptr, const bf16* __restrict__ Wb,
    const float* __restrict__ rowScale, void* __restrict__ Out,
    float* __restrict__ statSum, float* __restrict__ statSqs, int N)
{
    extern __shared__ bf16 wlds[];   // [DIM * LDSP], dynamic LDS -> offset 0

    if (threadIdx.x < 32) {
        unsigned long long ga = (unsigned long long)Wb;
        u32x4 g0;
        g0[0] = 1u;                               // count=1, no gather
        g0[1] = 0u;                               // lds_addr = 0 (dynamic LDS base)
        g0[2] = (unsigned int)ga;                 // global_addr[31:0]
        g0[3] = (unsigned int)(ga >> 32) | 0x80000000u;  // addr[56:32] | type=2
        i32x8 g1;
        g1[0] = (1 << 16)   // data_size = 2 bytes
              | (1 << 20)   // pad_enable
              | (6 << 22)   // pad_interval: every 128 DWORDs (one 512B row)
              | (3 << 25);  // pad_amount:   4 DWORDs (16B)
        g1[1] = (int)(256u << 16);  // tensor_dim0 = 256 (low 16 in [63:48])
        g1[2] = (int)(256u << 16);  // tensor_dim0 hi=0, tensor_dim1 = 256 (low 16)
        g1[3] = (int)(256u << 16);  // tensor_dim1 hi=0, tile_dim0 = 256
        g1[4] = 256;                // tile_dim1 = 256, tile_dim2 = 0
        g1[5] = 256;                // tensor_dim0_stride = 256 (low 32)
        g1[6] = 0;                  // stride0 hi, stride1 low
        g1[7] = 0;
        i32x4 gz4 = { 0, 0, 0, 0 };       // groups 2/3 unused (tile_dim3/4=0)
        i32x8 gz8 = { 0, 0, 0, 0, 0, 0, 0, 0 };
        __builtin_amdgcn_tensor_load_to_lds(g0, g1, gz4, gz4, gz8, 0);
        __builtin_amdgcn_s_wait_tensorcnt(0);
    }
    __syncthreads();

    const int tid  = threadIdx.x;
    const int wave = tid >> 5;
    const int lane = tid & 31;
    const int lhi  = lane >> 4;          // 0 / 1
    const int l15  = lane & 15;
    const int m0   = blockIdx.x * 64 + (wave & 3) * 16;
    const int jb   = (wave >> 2) * 128;
    if (m0 >= N) return;                 // wave-uniform, no barriers follow

    const int arow = (m0 + l15 < N) ? (m0 + l15) : (N - 1);

    v8f acc[8];
#pragma unroll
    for (int i = 0; i < 8; ++i)
#pragma unroll
        for (int r = 0; r < 8; ++r) acc[i][r] = 0.0f;

    for (int k0 = 0; k0 < DIM; k0 += 32) {
        const int kb = k0 + lhi * 8;
        v16bf a;
        if (A_F32) {
            const float* Af = (const float*)Aptr + (size_t)arow * DIM + kb;
            f32x4 f0 = *(const f32x4*)(Af);
            f32x4 f1 = *(const f32x4*)(Af + 4);
            f32x4 f2 = *(const f32x4*)(Af + 16);
            f32x4 f3 = *(const f32x4*)(Af + 20);
#pragma unroll
            for (int i = 0; i < 4; ++i) {
                a[i]      = (bf16)f0[i];
                a[4 + i]  = (bf16)f1[i];
                a[8 + i]  = (bf16)f2[i];
                a[12 + i] = (bf16)f3[i];
            }
        } else {
            const bf16* Ab = (const bf16*)Aptr + (size_t)arow * DIM + kb;
            bf16x8 b0 = *(const bf16x8*)Ab;
            bf16x8 b1 = *(const bf16x8*)(Ab + 16);
#pragma unroll
            for (int i = 0; i < 8; ++i) { a[i] = b0[i]; a[8 + i] = b1[i]; }
        }
#pragma unroll
        for (int nt = 0; nt < 8; ++nt) {
            const bf16* wr = &wlds[(jb + nt * 16 + l15) * LDSP + kb];
            bf16x8 w0 = *(const bf16x8*)wr;
            bf16x8 w1 = *(const bf16x8*)(wr + 16);
            v16bf b;
#pragma unroll
            for (int i = 0; i < 8; ++i) { b[i] = w0[i]; b[8 + i] = w1[i]; }
            acc[nt] = __builtin_amdgcn_wmma_f32_16x16x32_bf16(
                false, a, false, b, (short)0, acc[nt], false, false);
        }
    }

    // Epilogue: row scale, store, fused column statistics.
    float rsc[8];
#pragma unroll
    for (int r = 0; r < 8; ++r) {
        int row = m0 + r + lhi * 8;
        rsc[r] = ROW_SCALE ? rowScale[row < N ? row : 0] : 1.0f;
    }
#pragma unroll
    for (int nt = 0; nt < 8; ++nt) {
        const int j = jb + nt * 16 + l15;
        float cs = 0.0f, cq = 0.0f;
#pragma unroll
        for (int r = 0; r < 8; ++r) {
            int row = m0 + r + lhi * 8;
            if (row < N) {
                float v = acc[nt][r] * rsc[r];
                if (OUT_F32) ((float*)Out)[(size_t)row * DIM + j] = v;
                else         ((bf16*)Out)[(size_t)row * DIM + j] = (bf16)v;
                cs += v; cq += v * v;
            }
        }
        cs += __shfl_xor(cs, 16, 32);   // combine M-halves (lanes l and l+16)
        cq += __shfl_xor(cq, 16, 32);
        if (lane < 16) {
            atomicAdd(&statSum[j], cs);
            atomicAdd(&statSqs[j], cq);
        }
    }
}

// ---------------------------------------------------------------------------
// K4: in-place BN1 + ReLU on bf16 h
// ---------------------------------------------------------------------------
__global__ __launch_bounds__(256) void k4_bn_relu(
    bf16* __restrict__ h, const float* __restrict__ sum,
    const float* __restrict__ sqs, const float* __restrict__ gamma,
    const float* __restrict__ beta, float invN, int total8)
{
    int t = blockIdx.x * blockDim.x + threadIdx.x;
    if (t >= total8) return;
    int j = (t * 8) & (DIM - 1);
    bf16x8 v = ((bf16x8*)h)[t];
    bf16x8 o;
#pragma unroll
    for (int i = 0; i < 8; ++i) {
        float m   = sum[j + i] * invN;
        float var = sqs[j + i] * invN - m * m;
        float y = ((float)v[i] - m) * rsqrtf(var + BNEPS) * gamma[j + i] + beta[j + i];
        o[i] = (bf16)(y > 0.0f ? y : 0.0f);
    }
    ((bf16x8*)h)[t] = o;
}

// ---------------------------------------------------------------------------
// K6: out = relu(BN2(out)) + x, in-place fp32
// ---------------------------------------------------------------------------
__global__ __launch_bounds__(256) void k6_final(
    float* __restrict__ out, const float* __restrict__ x,
    const float* __restrict__ sum, const float* __restrict__ sqs,
    const float* __restrict__ gamma, const float* __restrict__ beta,
    float invN, int total4)
{
    int t = blockIdx.x * blockDim.x + threadIdx.x;
    if (t >= total4) return;
    int j = (t * 4) & (DIM - 1);
    f32x4 v  = ((f32x4*)out)[t];
    f32x4 xv = ((const f32x4*)x)[t];
#pragma unroll
    for (int i = 0; i < 4; ++i) {
        float m   = sum[j + i] * invN;
        float var = sqs[j + i] * invN - m * m;
        float y = (v[i] - m) * rsqrtf(var + BNEPS) * gamma[j + i] + beta[j + i];
        y = y > 0.0f ? y : 0.0f;
        v[i] = y + xv[i];
    }
    ((f32x4*)out)[t] = v;
}

// ---------------------------------------------------------------------------
extern "C" void kernel_launch(void* const* d_in, const int* in_sizes, int n_in,
                              void* d_out, int out_size, void* d_ws, size_t ws_size,
                              hipStream_t stream)
{
    (void)n_in; (void)out_size; (void)ws_size;
    const float* x      = (const float*)d_in[0];
    const int*   esrc   = (const int*)d_in[1];
    const int*   edst   = (const int*)d_in[2];
    const float* evals  = (const float*)d_in[3];
    const float* n_norm = (const float*)d_in[4];
    const float* w1     = (const float*)d_in[5];
    const float* gamma1 = (const float*)d_in[6];
    const float* beta1  = (const float*)d_in[7];
    const float* w2     = (const float*)d_in[8];
    const float* epsw   = (const float*)d_in[9];
    const float* gammah = (const float*)d_in[10];
    const float* betah  = (const float*)d_in[11];

    const int N = in_sizes[0] / DIM;
    const int E = in_sizes[1];

    char*  ws   = (char*)d_ws;
    float* agg  = (float*)ws;                              // N*DIM fp32
    size_t off  = (size_t)N * DIM * sizeof(float);
    bf16*  xb   = (bf16*)(ws + off);                       // N*DIM bf16 (x_bf, then h)
    bf16*  hbuf = xb;                                      // reused after edges done
    off        += (size_t)N * DIM * sizeof(bf16);
    float* stats = (float*)(ws + off);                     // 1024 floats
    float* sum1 = stats, *sqs1 = stats + 256, *sum2 = stats + 512, *sqs2 = stats + 768;
    off        += 4096;
    bf16*  w1b  = (bf16*)(ws + off);                       // 256x256 bf16
    off        += (size_t)DIM * DIM * sizeof(bf16);
    bf16*  w2b  = (bf16*)(ws + off);                       // 256x256 bf16

    const int thr    = 256;
    const int total8 = N * DIM / 8;
    const int total4 = N * DIM / 4;
    const float invN = 1.0f / (float)N;
    const size_t ldsb = (size_t)DIM * LDSP * sizeof(bf16); // 132 KB

    k0_init<<<(total8 + thr - 1) / thr, thr, 0, stream>>>(
        x, epsw, w1, w2, agg, xb, w1b, w2b, stats, total8);

    k1_edges<<<(E + 7) / 8, thr, 0, stream>>>(esrc, edst, evals, xb, agg, E);

    int mblocks = (N + 63) / 64;
    gemm_wmma<true, false, false><<<mblocks, thr, ldsb, stream>>>(
        agg, w1b, nullptr, hbuf, sum1, sqs1, N);

    k4_bn_relu<<<(total8 + thr - 1) / thr, thr, 0, stream>>>(
        hbuf, sum1, sqs1, gamma1, beta1, invN, total8);

    gemm_wmma<false, true, true><<<mblocks, thr, ldsb, stream>>>(
        hbuf, w2b, n_norm, d_out, sum2, sqs2, N);

    k6_final<<<(total4 + thr - 1) / thr, thr, 0, stream>>>(
        (float*)d_out, x, sum2, sqs2, gammah, betah, invN, total4);
}
